// MrLoraLayer_24386824307236
// MI455X (gfx1250) — matrix operand order repaired and checked by reading, not compiled, not run on hardware
//
#include <hip/hip_runtime.h>
#include <hip/hip_bf16.h>
#include <stdint.h>

// ---------------------------------------------------------------------------
// MrLoRA fused layer for gfx1250 (MI455X).
//   out = x @ W^T + b + sum_i coef_i * (x @ A_i^T) @ B_i^T
//
// Precision: 3-term bf16 split-precision emulation of fp32 GEMM:
//   x = xh + xl, w = wh + wl  ->  acc += xh*wh + xh*wl + xl*wh   (fp32 acc)
// Matrix ops: v_wmma_f32_16x16x32_bf16 throughout.
// Main GEMM: 128x128 block tile (dynamic LDS, 80 KB, hi/lo double-buffered),
// register-prefetch pipeline, one barrier per K-step, NT stores for `out`.
// ---------------------------------------------------------------------------

typedef __attribute__((ext_vector_type(16))) __bf16 v16bf;
typedef __attribute__((ext_vector_type(8)))  __bf16 v8bf;
typedef __attribute__((ext_vector_type(8)))  float  v8f;

#define D_IN   4096
#define D_OUT  4096
#define M_TOK  8192
#define RPAD   32

#define WMMA_BF16(a, b, c) \
  __builtin_amdgcn_wmma_f32_16x16x32_bf16(false, (a), false, (b), (short)0, (c), false, false)

// ---- helpers ---------------------------------------------------------------

__device__ __forceinline__ void split1(float f, __bf16& h, __bf16& l) {
  h = (__bf16)f;
  l = (__bf16)(f - (float)h);
}

// split 8 fp32 (two float4 register pairs) into hi/lo bf16 and store 16B each
__device__ __forceinline__ void split8st(__bf16* dh, __bf16* dl, float4 a, float4 b) {
  float f[8] = {a.x, a.y, a.z, a.w, b.x, b.y, b.z, b.w};
  v8bf h, l;
#pragma unroll
  for (int i = 0; i < 8; ++i) { __bf16 hh, ll; split1(f[i], hh, ll); h[i] = hh; l[i] = ll; }
  *(v8bf*)dh = h;
  *(v8bf*)dl = l;
}

__device__ __forceinline__ v8bf ld8(const __bf16* p) { return *(const v8bf*)p; }

__device__ __forceinline__ v16bf comb(v8bf lo, v8bf hi) {
  v16bf a;
#pragma unroll
  for (int i = 0; i < 8; ++i) { a[i] = lo[i]; a[8 + i] = hi[i]; }
  return a;
}

// A-operand fragment (16x32 bf16), ISA layout:
// lanes 0-15 hold K {0..7, 16..23}; lanes 16-31 hold K {8..15, 24..31}.
__device__ __forceinline__ v16bf fragA(const __bf16* row, int lane) {
  const int kb = (lane < 16) ? 0 : 8;
  return comb(ld8(row + kb), ld8(row + kb + 16));
}
// B-operand fragment (32x16 bf16): lanes 0-15 hold K {0..7, 16..23}; lanes 16-31 K {8..15, 24..31}.
__device__ __forceinline__ v16bf fragB(const __bf16* row, int lane) {
  const int kb = (lane < 16) ? 0 : 8;
  return comb(ld8(row + kb), ld8(row + kb + 16));
}
// A fragment hi/lo directly from an fp32 row in global memory.
__device__ __forceinline__ void fragA_split(const float* row, int lane,
                                            v16bf& hi, v16bf& lo) {
  const int kb = (lane < 16) ? 0 : 8;
#pragma unroll
  for (int g = 0; g < 2; ++g) {
    const float4* q = (const float4*)(row + kb + g * 16);
    float4 a = q[0], b = q[1];
    float f[8] = {a.x, a.y, a.z, a.w, b.x, b.y, b.z, b.w};
#pragma unroll
    for (int i = 0; i < 8; ++i) {
      __bf16 h, l; split1(f[i], h, l);
      hi[g * 8 + i] = h; lo[g * 8 + i] = l;
    }
  }
}

// ---- kernel 0: pack A_cat (hi/lo) and scaled B_cat (hi/lo) -----------------

__global__ __launch_bounds__(256) void pack_kernel(
    const float* __restrict__ A16, const float* __restrict__ B16,
    const float* __restrict__ A8,  const float* __restrict__ B8,
    const float* __restrict__ A4,  const float* __restrict__ B4,
    const float* __restrict__ A2,  const float* __restrict__ B2,
    const float* __restrict__ A1,  const float* __restrict__ B1,
    const float* __restrict__ alphas,
    __bf16* __restrict__ Ah, __bf16* __restrict__ Al,
    __bf16* __restrict__ Bh, __bf16* __restrict__ Bl) {
  int idx = blockIdx.x * 256 + threadIdx.x;
  if (idx < RPAD * D_IN) {                       // A_cat[j][k], j = rank slot
    int j = idx >> 12, k = idx & 4095;
    float v = 0.0f;
    if      (j < 16) v = A16[j * D_IN + k];
    else if (j < 24) v = A8[(j - 16) * D_IN + k];
    else if (j < 28) v = A4[(j - 24) * D_IN + k];
    else if (j < 30) v = A2[(j - 28) * D_IN + k];
    else if (j == 30) v = A1[k];
    __bf16 h, l; split1(v, h, l);
    Ah[idx] = h; Al[idx] = l;
  } else {                                       // B_cat[o][j], scale folded in
    int id2 = idx - RPAD * D_IN;
    if (id2 < D_OUT * RPAD) {
      int o = id2 >> 5, j = id2 & 31;
      float v = 0.0f;
      if      (j < 16) v = alphas[0] * 4.0f                * B16[o * 16 + j];
      else if (j < 24) v = alphas[1] * 5.656854249492381f  * B8[o * 8 + (j - 16)];
      else if (j < 28) v = alphas[2] * 8.0f                * B4[o * 4 + (j - 24)];
      else if (j < 30) v = alphas[3] * 11.313708498984761f * B2[o * 2 + (j - 28)];
      else if (j == 30) v = alphas[4] * 16.0f              * B1[o];
      __bf16 h, l; split1(v, h, l);
      Bh[(size_t)o * RPAD + j] = h; Bl[(size_t)o * RPAD + j] = l;
    }
  }
}

// ---- kernel 1: t = x @ A_cat^T (split precision), t stored as hi/lo --------
// One wave computes a 16x32 tile of t; 6 WMMAs per K-step.

__global__ __launch_bounds__(256) void lora_t_kernel(
    const float* __restrict__ x,
    const __bf16* __restrict__ Ah, const __bf16* __restrict__ Al,
    __bf16* __restrict__ th, __bf16* __restrict__ tl) {
  const int lane = threadIdx.x & 31;
  const int wid  = threadIdx.x >> 5;
  const int m0   = (blockIdx.x * 8 + wid) * 16;
  const int frow = lane & 15;

  const float* xrow = x + (size_t)(m0 + frow) * D_IN;
  const size_t ar0  = (size_t)frow * D_IN;        // rank slots 0..15
  const size_t ar1  = (size_t)(16 + frow) * D_IN; // rank slots 16..31

  v8f c0 = {}, c1 = {};
  for (int k0 = 0; k0 < D_IN; k0 += 32) {
    __builtin_prefetch(xrow + k0 + 128, 0, 0);
    v16bf ah, al;
    fragA_split(xrow + k0, lane, ah, al);
    v16bf b0h = fragB(Ah + ar0 + k0, lane);
    v16bf b0l = fragB(Al + ar0 + k0, lane);
    v16bf b1h = fragB(Ah + ar1 + k0, lane);
    v16bf b1l = fragB(Al + ar1 + k0, lane);
    c0 = WMMA_BF16(ah, b0h, c0);
    c1 = WMMA_BF16(ah, b1h, c1);
    c0 = WMMA_BF16(ah, b0l, c0);
    c1 = WMMA_BF16(ah, b1l, c1);
    c0 = WMMA_BF16(al, b0h, c0);
    c1 = WMMA_BF16(al, b1h, c1);
  }
  // C layout: VGPR i -> M = i + (lane<16 ? 0 : 8), N = lane%16.
  const int mb = m0 + ((lane < 16) ? 0 : 8);
#pragma unroll
  for (int i = 0; i < 8; ++i) {
    __bf16 h, l;
    split1(c0[i], h, l);
    th[(size_t)(mb + i) * RPAD + frow] = h;
    tl[(size_t)(mb + i) * RPAD + frow] = l;
    split1(c1[i], h, l);
    th[(size_t)(mb + i) * RPAD + 16 + frow] = h;
    tl[(size_t)(mb + i) * RPAD + 16 + frow] = l;
  }
}

// ---- kernel 2: main fused GEMM ---------------------------------------------
// Block: 256 thr (8 waves), tile 128(M) x 128(N), BK = 32.
// Dynamic LDS: hi/lo x/W tiles, double-buffered (80 KB total).
// Wave grid 4(M) x 2(N); each wave owns a 32x64 sub-tile = 2x4 accumulators,
// 24 WMMAs per K-step.

#define BM 128
#define BN 128
#define BK 32
#define LDSTR 40            // padded bf16 row stride (80 B) -> conflict-free
#define TILE_E (BM * LDSTR) // 5120 bf16 elements per tile buffer

__global__ __launch_bounds__(256) void gemm_kernel(
    const float* __restrict__ x, const float* __restrict__ W,
    const float* __restrict__ bias,
    const __bf16* __restrict__ th, const __bf16* __restrict__ tl,
    const __bf16* __restrict__ bch, const __bf16* __restrict__ bcl,
    float* __restrict__ out) {
  extern __shared__ __bf16 smem[];   // 8 * TILE_E elements = 80 KB
  __bf16* sxh = smem;                // [2][TILE_E]
  __bf16* sxl = smem + 2 * TILE_E;   // [2][TILE_E]
  __bf16* swh = smem + 4 * TILE_E;   // [2][TILE_E]
  __bf16* swl = smem + 6 * TILE_E;   // [2][TILE_E]

  const int n0   = blockIdx.x * BN;
  const int m0   = blockIdx.y * BM;
  const int tid  = threadIdx.x;
  const int lane = tid & 31;
  const int wid  = tid >> 5;
  const int wm   = wid & 3;   // 0..3 -> 32-row strip
  const int wn   = wid >> 2;  // 0..1 -> 64-col strip
  const int frow = lane & 15;

  // staging: 2 threads per row, 16 floats each (for both x and W tiles)
  const int lr = tid >> 1, lc = tid & 1;
  const float* xrow = x + (size_t)(m0 + lr) * D_IN + lc * 16;
  const float* wrow = W + (size_t)(n0 + lr) * D_IN + lc * 16;
  const int soff = lr * LDSTR + lc * 16;

  float4 px[4], pw[4];
  auto ldtile = [&](int k0) {
#pragma unroll
    for (int j = 0; j < 4; ++j) px[j] = *(const float4*)(xrow + k0 + 4 * j);
#pragma unroll
    for (int j = 0; j < 4; ++j) pw[j] = *(const float4*)(wrow + k0 + 4 * j);
  };
  auto sttile = [&](int buf) {
    const int o = buf * TILE_E + soff;
    split8st(&sxh[o],     &sxl[o],     px[0], px[1]);
    split8st(&sxh[o + 8], &sxl[o + 8], px[2], px[3]);
    split8st(&swh[o],     &swl[o],     pw[0], pw[1]);
    split8st(&swh[o + 8], &swl[o + 8], pw[2], pw[3]);
  };

  v8f acc[2][4];
#pragma unroll
  for (int mt = 0; mt < 2; ++mt)
#pragma unroll
    for (int nt = 0; nt < 4; ++nt) acc[mt][nt] = (v8f){};

  // prologue: stage tile 0
  ldtile(0);
  sttile(0);
  __syncthreads();

  int cur = 0;
  for (int k0 = 0; k0 < D_IN; k0 += BK) {
    const bool more = (k0 + BK) < D_IN;
    if (more) {
      __builtin_prefetch(xrow + k0 + 2 * BK, 0, 0);
      __builtin_prefetch(wrow + k0 + 2 * BK, 0, 0);
      ldtile(k0 + BK);
    }

    v16bf ah[2], al[2];
#pragma unroll
    for (int mt = 0; mt < 2; ++mt) {
      const int r = cur * TILE_E + (wm * 32 + mt * 16 + frow) * LDSTR;
      ah[mt] = fragA(&sxh[r], lane);
      al[mt] = fragA(&sxl[r], lane);
    }
#pragma unroll
    for (int nt = 0; nt < 4; ++nt) {
      const int r = cur * TILE_E + (wn * 64 + nt * 16 + frow) * LDSTR;
      v16bf bh = fragB(&swh[r], lane);
      v16bf bl = fragB(&swl[r], lane);
#pragma unroll
      for (int mt = 0; mt < 2; ++mt) {
        acc[mt][nt] = WMMA_BF16(ah[mt], bh, acc[mt][nt]);
        acc[mt][nt] = WMMA_BF16(ah[mt], bl, acc[mt][nt]);
        acc[mt][nt] = WMMA_BF16(al[mt], bh, acc[mt][nt]);
      }
    }

    if (more) sttile(cur ^ 1);
    __syncthreads();
    cur ^= 1;
  }

  // LoRA epilogue: rank-32 update, split precision (3 WMMAs per tile).
  v16bf tah[2], tal[2];
#pragma unroll
  for (int mt = 0; mt < 2; ++mt) {
    const size_t r = (size_t)(m0 + wm * 32 + mt * 16 + frow) * RPAD;
    tah[mt] = fragA(th + r, lane);
    tal[mt] = fragA(tl + r, lane);
  }
#pragma unroll
  for (int nt = 0; nt < 4; ++nt) {
    const size_t r = (size_t)(n0 + wn * 64 + nt * 16 + frow) * RPAD;
    v16bf bh = fragB(bch + r, lane);
    v16bf bl = fragB(bcl + r, lane);
#pragma unroll
    for (int mt = 0; mt < 2; ++mt) {
      acc[mt][nt] = WMMA_BF16(tah[mt], bh, acc[mt][nt]);
      acc[mt][nt] = WMMA_BF16(tah[mt], bl, acc[mt][nt]);
      acc[mt][nt] = WMMA_BF16(tal[mt], bh, acc[mt][nt]);
    }
  }

  // bias + store (C layout: VGPR i -> M = i + (lane<16?0:8), N = lane%16)
  // out is written once and never re-read -> non-temporal stores.
#pragma unroll
  for (int nt = 0; nt < 4; ++nt) {
    const int col = n0 + wn * 64 + nt * 16 + frow;
    const float bv = bias[col];
#pragma unroll
    for (int mt = 0; mt < 2; ++mt) {
      const int rbase = m0 + wm * 32 + mt * 16 + ((lane < 16) ? 0 : 8);
#pragma unroll
      for (int i = 0; i < 8; ++i)
        __builtin_nontemporal_store(acc[mt][nt][i] + bv,
                                    &out[(size_t)(rbase + i) * D_OUT + col]);
    }
  }
}

// ---- host-side launch ------------------------------------------------------

extern "C" void kernel_launch(void* const* d_in, const int* in_sizes, int n_in,
                              void* d_out, int out_size, void* d_ws, size_t ws_size,
                              hipStream_t stream) {
  (void)in_sizes; (void)n_in; (void)out_size; (void)ws_size;
  const float* x      = (const float*)d_in[0];
  const float* W      = (const float*)d_in[1];
  const float* b      = (const float*)d_in[2];
  const float* A16    = (const float*)d_in[3];
  const float* B16    = (const float*)d_in[4];
  const float* A8     = (const float*)d_in[5];
  const float* B8     = (const float*)d_in[6];
  const float* A4     = (const float*)d_in[7];
  const float* B4     = (const float*)d_in[8];
  const float* A2     = (const float*)d_in[9];
  const float* B2     = (const float*)d_in[10];
  const float* A1     = (const float*)d_in[11];
  const float* B1     = (const float*)d_in[12];
  const float* alphas = (const float*)d_in[13];
  float* out = (float*)d_out;

  const size_t SZ_A = (size_t)RPAD * D_IN * 2;   // 256 KB each
  const size_t SZ_B = (size_t)D_OUT * RPAD * 2;  // 256 KB each
  const size_t SZ_T = (size_t)M_TOK * RPAD * 2;  // 512 KB each
  char* wsb = (char*)d_ws;
  __bf16* Ah = (__bf16*)(wsb);
  __bf16* Al = (__bf16*)(wsb + SZ_A);
  __bf16* Bh = (__bf16*)(wsb + 2 * SZ_A);
  __bf16* Bl = (__bf16*)(wsb + 2 * SZ_A + SZ_B);
  __bf16* th = (__bf16*)(wsb + 2 * SZ_A + 2 * SZ_B);
  __bf16* tlo = (__bf16*)(wsb + 2 * SZ_A + 2 * SZ_B + SZ_T);

  hipLaunchKernelGGL(pack_kernel, dim3(1024), dim3(256), 0, stream,
                     A16, B16, A8, B8, A4, B4, A2, B2, A1, B1, alphas,
                     Ah, Al, Bh, Bl);
  hipLaunchKernelGGL(lora_t_kernel, dim3(64), dim3(256), 0, stream,
                     x, Ah, Al, th, tlo);
  const size_t lds_bytes = (size_t)8 * TILE_E * sizeof(__bf16);  // 80 KB
  hipLaunchKernelGGL(gemm_kernel, dim3(D_OUT / BN, M_TOK / BM), dim3(256),
                     lds_bytes, stream, x, W, b, th, tlo, Bh, Bl, out);
}
